// LossMeanCov_34230889349412
// MI455X (gfx1250) — matrix operand order repaired and checked by reading, not compiled, not run on hardware
//
#include <hip/hip_runtime.h>
#include <hip/hip_bf16.h>

typedef __attribute__((ext_vector_type(16))) _Float16 v16h;
typedef __attribute__((ext_vector_type(8)))  float    v8f;

#define N_PTS     131072
#define DIMS      32
#define KCLUST    64
#define BETA_     10.0f
#define TILE_ROWS 256
#define THREADS   512                    /* 16 waves -> 16x16-row WMMA tiles */
#define NTILES    (N_PTS / TILE_ROWS)    /* 512 */
#define GRID      256

/* workspace (float) layout */
#define WS_FILL 0
#define WS_CNT  64
#define WS_SUM  128           /* 64*32  */
#define WS_M2   2176          /* 64*32*32, upper triangle valid */
#define WS_TOT  67712

/* dynamic LDS (floats): cnt[64]|sum[2048]|cc[64]|x[8192]|m2[65536]
   = 75904 floats = 303616 B  (< 320 KB CDNA5 WGP LDS) */
#define SM_CNT 0
#define SM_SUM 64
#define SM_CC  2112
#define SM_X   2176
#define SM_M2  10368
#define SMEM_FLOATS (SM_M2 + 65536)

#if __has_builtin(__builtin_amdgcn_global_load_async_to_lds_b128)
#define HAVE_ASYNC_COPY 1
typedef int v4i_ __attribute__((vector_size(4 * sizeof(int))));
typedef __attribute__((address_space(1))) v4i_ gas_v4i;   /* global int4 */
typedef __attribute__((address_space(3))) v4i_ las_v4i;   /* LDS int4 */
#endif

#if __has_builtin(__builtin_amdgcn_permlane16)
#define HAVE_PERMLANE16 1
/* xor-butterfly selectors within each 16-lane group (nibble i = i^mask) */
#define PL16(v, S0, S1) __builtin_amdgcn_permlane16((v), (v), (S0), (S1), false, false)

__device__ __forceinline__ unsigned long long kmin_step(unsigned long long k, int s0, int s1) {
    int lo = PL16((int)(unsigned)(k & 0xffffffffull), s0, s1);
    int hi = PL16((int)(unsigned)(k >> 32), s0, s1);
    unsigned long long o = ((unsigned long long)(unsigned)hi << 32) | (unsigned)lo;
    return o < k ? o : k;
}
__device__ __forceinline__ float fsum_step(float z, int s0, int s1) {
    return z + __int_as_float(PL16(__float_as_int(z), s0, s1));
}
#endif

/* min-reduce packed (d2,idx) key over the 16-lane group */
__device__ __forceinline__ unsigned long long kmin16(unsigned long long k) {
#ifdef HAVE_PERMLANE16
    k = kmin_step(k, 0x67452301, 0xEFCDAB89);   /* xor 1 */
    k = kmin_step(k, 0x54761032, 0xDCFE98BA);   /* xor 2 */
    k = kmin_step(k, 0x32107654, 0xBA98FEDC);   /* xor 4 */
    k = kmin_step(k, 0xFEDCBA98, 0x76543210);   /* xor 8 */
#else
    for (int m = 8; m >= 1; m >>= 1) {
        unsigned long long o = (unsigned long long)__shfl_xor((long long)k, m, 32);
        k = o < k ? o : k;
    }
#endif
    return k;
}
__device__ __forceinline__ float fsum16(float z) {
#ifdef HAVE_PERMLANE16
    z = fsum_step(z, 0x67452301, 0xEFCDAB89);
    z = fsum_step(z, 0x54761032, 0xDCFE98BA);
    z = fsum_step(z, 0x32107654, 0xBA98FEDC);
    z = fsum_step(z, 0xFEDCBA98, 0x76543210);
#else
    for (int m = 8; m >= 1; m >>= 1) z += __shfl_xor(z, m, 32);
#endif
    return z;
}

__device__ __forceinline__ float fast_rcp(float v) {
#if __has_builtin(__builtin_amdgcn_rcpf)
    return __builtin_amdgcn_rcpf(v);   /* single v_rcp_f32 */
#else
    return 1.0f / v;
#endif
}

__global__ void __launch_bounds__(256) zero_ws_kernel(float* __restrict__ ws) {
    int i = blockIdx.x * blockDim.x + threadIdx.x;
    if (i < WS_TOT) ws[i] = 0.0f;
}

__global__ void __launch_bounds__(THREADS) cluster_pass_kernel(
    const float* __restrict__ x, const float* __restrict__ centers, float* __restrict__ ws)
{
    extern __shared__ float smem[];
    float* s_cnt = smem + SM_CNT;
    float* s_sum = smem + SM_SUM;
    float* s_cc  = smem + SM_CC;
    float* s_x   = smem + SM_X;
    float* s_m2  = smem + SM_M2;

    const int tid  = threadIdx.x;
    const int lane = tid & 31;
    const int wave = tid >> 5;
    const int sub  = lane & 15;   /* cluster-in-tile column / A-row in wave tile */
    const int g    = lane >> 4;   /* K half-group of the 16-bit WMMA layout */

    for (int i = tid; i < 2112; i += THREADS) smem[i] = 0.0f;      /* cnt|sum */
    for (int i = tid; i < 65536; i += THREADS) s_m2[i] = 0.0f;
    for (int c = tid; c < KCLUST; c += THREADS) {
        float s = 0.0f;
        for (int k = 0; k < DIMS; ++k) { float v = centers[c * DIMS + k]; s += v * v; }
        s_cc[c] = s;
    }
    __syncthreads();

    float ccr[4];
#pragma unroll
    for (int t = 0; t < 4; ++t) ccr[t] = s_cc[t * 16 + sub];

    /* register-resident soft-occupancy accumulators: this lane always owns
       cluster columns t*16+sub for the whole kernel */
    float fillacc[4] = {0.0f, 0.0f, 0.0f, 0.0f};

    /* B fragments: 4 tiles x 16 clusters, K=32 */
    v16h bfrag[4];
#pragma unroll
    for (int t = 0; t < 4; ++t) {
        const float* cp = centers + (t * 16 + sub) * DIMS + g * 16;
#pragma unroll
        for (int e = 0; e < 16; ++e) bfrag[t][e] = (_Float16)cp[e];
    }

    for (int tile = blockIdx.x; tile < NTILES; tile += gridDim.x) {
        const int rowBase = tile * TILE_ROWS;

        if (tile + (int)gridDim.x < NTILES) {
            const float* nxt = x + (size_t)(tile + gridDim.x) * TILE_ROWS * DIMS;
            __builtin_prefetch(nxt + tid * 16, 0, 0);   /* global_prefetch_b8 */
        }

        /* stage x tile in LDS (async global->LDS on CDNA5) */
        {
            const float4* xg = (const float4*)(x + (size_t)rowBase * DIMS);
            float4* xs = (float4*)s_x;
#ifdef HAVE_ASYNC_COPY
            for (int i = tid; i < TILE_ROWS * DIMS / 4; i += THREADS)
                __builtin_amdgcn_global_load_async_to_lds_b128(
                    (gas_v4i*)(xg + i), (las_v4i*)(xs + i), 0, 0);
#if __has_builtin(__builtin_amdgcn_s_wait_asynccnt)
            __builtin_amdgcn_s_wait_asynccnt(0);
#else
            asm volatile("s_wait_asynccnt 0x0" ::: "memory");
#endif
#else
            for (int i = tid; i < TILE_ROWS * DIMS / 4; i += THREADS) xs[i] = xg[i];
#endif
        }
        __syncthreads();

        /* A fragment: wave's 16 rows; lane row = sub; half g holds K = 16g+0..15 */
        v16h afrag;
        {
            const float* xp = s_x + (wave * 16 + sub) * DIMS;
#pragma unroll
            for (int e = 0; e < 8; ++e) afrag[e]     = (_Float16)xp[16 * g + e];
#pragma unroll
            for (int e = 0; e < 8; ++e) afrag[8 + e] = (_Float16)xp[16 * g + 8 + e];
        }

        /* all 64 centroid dot-products: 4x v_wmma_f32_16x16x32_f16 */
        v8f acc[4];
#pragma unroll
        for (int t = 0; t < 4; ++t) {
            v8f z = {};
            acc[t] = __builtin_amdgcn_wmma_f32_16x16x32_f16(
                false, afrag, false, bfrag[t], (short)0, z, false, false);
        }

        /* C layout: lane holds column N=sub; acc element r is row M = 8g + r. */
#pragma unroll 1
        for (int r = 0; r < 8; ++r) {
            const int rloc = wave * 16 + 8 * g + r;

            /* --- phase 1: all DS loads for this row, issued together --- */
            float xa_[DIMS];
            {
                const float4* xr4 = (const float4*)(s_x + rloc * DIMS);
#pragma unroll
                for (int i = 0; i < 8; ++i) {
                    const float4 q = xr4[i];
                    xa_[4 * i + 0] = q.x; xa_[4 * i + 1] = q.y;
                    xa_[4 * i + 2] = q.z; xa_[4 * i + 3] = q.w;
                }
            }
            const float xb0 = s_x[rloc * DIMS + sub];
            const float xb1 = s_x[rloc * DIMS + sub + 16];

            /* --- phase 2: pure-VALU math + permlane reductions --- */
            float xxr = 0.0f;
#pragma unroll
            for (int a = 0; a < DIMS; ++a) xxr += xa_[a] * xa_[a];

            float d2[4];
#pragma unroll
            for (int t = 0; t < 4; ++t) d2[t] = xxr - 2.0f * acc[t][r] + ccr[t];

            /* per-lane min over 4 tiles with first-index tie-break */
            float mn4 = d2[0]; int ci4 = sub;
#pragma unroll
            for (int t = 1; t < 4; ++t)
                if (d2[t] < mn4) { mn4 = d2[t]; ci4 = t * 16 + sub; }

            /* fused min+argmin across the 16-lane group: packed (d2_bits<<6)|idx */
            unsigned long long key =
                (((unsigned long long)__float_as_uint(fmaxf(mn4, 0.0f))) << 6) | (unsigned)ci4;
            key = kmin16(key);
            const int   c  = (int)(key & 63ull);
            const float mn = __uint_as_float((unsigned)(key >> 6));

            float e4[4]; float z = 0.0f;
#pragma unroll
            for (int t = 0; t < 4; ++t) { e4[t] = __expf(-BETA_ * (d2[t] - mn)); z += e4[t]; }
            z = fsum16(z);
            const float inv = fast_rcp(z);     /* v_rcp_f32, not IEEE div expansion */
#pragma unroll
            for (int t = 0; t < 4; ++t) fillacc[t] += e4[t] * inv;   /* register accumulate */

            /* --- phase 3: uninterrupted ds_add_f32 stream --- */
            if (sub == 0) atomicAdd(&s_cnt[c], 1.0f);
            atomicAdd(&s_sum[c * DIMS + sub],      xb0);
            atomicAdd(&s_sum[c * DIMS + sub + 16], xb1);

            float* m2c = s_m2 + c * DIMS * DIMS;   /* symmetric: upper triangle only */
#pragma unroll
            for (int a = 0; a < 16; ++a)
                if (a <= sub) atomicAdd(&m2c[a * DIMS + sub], xa_[a] * xb0);
#pragma unroll
            for (int a = 0; a < DIMS; ++a)
                if (a <= sub + 16) atomicAdd(&m2c[a * DIMS + sub + 16], xa_[a] * xb1);
        }
        __syncthreads();   /* protect s_x before next tile overwrite */
    }

    /* flush: register fill accumulators + LDS cnt|sum + m2 upper triangle */
#pragma unroll
    for (int t = 0; t < 4; ++t) atomicAdd(&ws[WS_FILL + t * 16 + sub], fillacc[t]);
    for (int i = tid; i < 2112; i += THREADS) atomicAdd(&ws[WS_CNT + i], smem[i]);
    for (int i = tid; i < 65536; i += THREADS) {
        const int a = (i >> 5) & 31, b = i & 31;
        if (a <= b) atomicAdd(&ws[WS_M2 + i], s_m2[i]);
    }
}

__global__ void __launch_bounds__(256) finalize_kernel(
    const float* __restrict__ ws, const float* __restrict__ ft,
    const float* __restrict__ mt, const float* __restrict__ ct, float* __restrict__ out)
{
    __shared__ float s_mean[KCLUST * DIMS];
    __shared__ float s_dnm[KCLUST];
    __shared__ float red[256];
    const int tid = threadIdx.x;

    for (int i = tid; i < KCLUST; i += 256) s_dnm[i] = fmaxf(ws[WS_CNT + i], 1.0f);
    __syncthreads();
    for (int i = tid; i < KCLUST * DIMS; i += 256) s_mean[i] = ws[WS_SUM + i] / s_dnm[i >> 5];
    __syncthreads();

    float part = 0.0f;
    const float invN = 1.0f / (float)N_PTS;
    for (int i = tid; i < KCLUST; i += 256) {
        float d = ws[WS_FILL + i] * invN - ft[i];
        part += d * d * (1.0f / (float)KCLUST);
    }
    for (int i = tid; i < KCLUST * DIMS; i += 256) {
        float d = s_mean[i] - mt[i];
        part += d * d * (1.0f / (float)(KCLUST * DIMS));
    }
    for (int i = tid; i < KCLUST * DIMS * DIMS; i += 256) {
        const int c = i >> 10, a = (i >> 5) & 31, b = i & 31;
        const int j = (a <= b) ? i : ((c << 10) + (b << 5) + a);   /* mirror lower triangle */
        float cov = ws[WS_M2 + j] / s_dnm[c] - s_mean[(c << 5) + a] * s_mean[(c << 5) + b];
        float d = cov - ct[i];
        part += d * d * (1.0f / (float)(KCLUST * DIMS * DIMS));
    }
    red[tid] = part;
    __syncthreads();
    for (int s = 128; s > 0; s >>= 1) {
        if (tid < s) red[tid] += red[tid + s];
        __syncthreads();
    }
    if (tid == 0) out[0] = red[0];
}

extern "C" void kernel_launch(void* const* d_in, const int* in_sizes, int n_in,
                              void* d_out, int out_size, void* d_ws, size_t ws_size,
                              hipStream_t stream) {
    (void)in_sizes; (void)n_in; (void)out_size; (void)ws_size;
    const float* x       = (const float*)d_in[0];
    const float* centers = (const float*)d_in[1];
    const float* ft      = (const float*)d_in[2];
    const float* mt      = (const float*)d_in[3];
    const float* ct      = (const float*)d_in[4];
    float* ws  = (float*)d_ws;
    float* out = (float*)d_out;

    zero_ws_kernel<<<(WS_TOT + 255) / 256, 256, 0, stream>>>(ws);
    cluster_pass_kernel<<<GRID, THREADS, SMEM_FLOATS * sizeof(float), stream>>>(x, centers, ws);
    finalize_kernel<<<1, 256, 0, stream>>>(ws, ft, mt, ct, out);
}
